// PatchCoreCLIP_70042326663330
// MI455X (gfx1250) — matrix operand order repaired and checked by reading, not compiled, not run on hardware
//
#include <hip/hip_runtime.h>
#include <hip/hip_bf16.h>

typedef __attribute__((ext_vector_type(16))) _Float16 v16h;
typedef __attribute__((ext_vector_type(8)))  _Float16 v8h;
typedef __attribute__((ext_vector_type(8)))  float    v8f;

#define D_DIM   768
#define NPATCH  3136      // 64 images * 49 patches
#define MBANK   100000
#define PATCHES_PER_BLK 32   // 2 WMMA A-tiles per block
#define CHUNK   2048         // bank columns per block
#define KSTEPS  24           // 768 / 32
#define APITCH  776          // 768 + 8 f16 pad -> LDS rows land on distinct banks

// ---------------------------------------------------------------------------
// Stage 1: drop CLS, L2-normalize patches -> f16 A [3136 x 768], p2, init min
// ---------------------------------------------------------------------------
__global__ __launch_bounds__(256) void prep_patches_kernel(
    const float* __restrict__ tokens, _Float16* __restrict__ Ah,
    float* __restrict__ p2, int* __restrict__ mind2)
{
  const int q = blockIdx.x;            // flat patch id
  const int b = q / 49, p = q % 49;
  const float* src = tokens + (long long)(b * 50 + p + 1) * D_DIM;
  const int t = threadIdx.x;

  float x0 = src[t], x1 = src[t + 256], x2 = src[t + 512];
  float s = x0 * x0 + x1 * x1 + x2 * x2;
  s += __shfl_xor(s, 16); s += __shfl_xor(s, 8);
  s += __shfl_xor(s, 4);  s += __shfl_xor(s, 2); s += __shfl_xor(s, 1);

  __shared__ float wsum[8];
  __shared__ float stot;
  if ((t & 31) == 0) wsum[t >> 5] = s;
  __syncthreads();
  if (t == 0) {
    float a = 0.f;
    for (int i = 0; i < 8; ++i) a += wsum[i];
    stot = a;
  }
  __syncthreads();
  const float tot = stot;
  const float rn  = 1.0f / (sqrtf(tot) + 1e-12f);

  _Float16* drow = Ah + (long long)q * D_DIM;
  drow[t]       = (_Float16)(x0 * rn);
  drow[t + 256] = (_Float16)(x1 * rn);
  drow[t + 512] = (_Float16)(x2 * rn);
  if (t == 0) { p2[q] = tot * rn * rn; mind2[q] = 0x7F800000; }
}

// ---------------------------------------------------------------------------
// Stage 2: L2-normalize memory bank -> f16 B [100000 x 768] row-major, m2
// ---------------------------------------------------------------------------
__global__ __launch_bounds__(256) void prep_bank_kernel(
    const float* __restrict__ bank, _Float16* __restrict__ Bh,
    float* __restrict__ m2)
{
  const long long row = blockIdx.x;
  const float* src = bank + row * D_DIM;
  const int t = threadIdx.x;

  float x0 = src[t], x1 = src[t + 256], x2 = src[t + 512];
  float s = x0 * x0 + x1 * x1 + x2 * x2;
  s += __shfl_xor(s, 16); s += __shfl_xor(s, 8);
  s += __shfl_xor(s, 4);  s += __shfl_xor(s, 2); s += __shfl_xor(s, 1);

  __shared__ float wsum[8];
  __shared__ float stot;
  if ((t & 31) == 0) wsum[t >> 5] = s;
  __syncthreads();
  if (t == 0) {
    float a = 0.f;
    for (int i = 0; i < 8; ++i) a += wsum[i];
    stot = a;
  }
  __syncthreads();
  const float tot = stot;
  const float rn  = 1.0f / (sqrtf(tot) + 1e-12f);

  _Float16* drow = Bh + row * D_DIM;
  drow[t]       = (_Float16)(x0 * rn);
  drow[t + 256] = (_Float16)(x1 * rn);
  drow[t + 512] = (_Float16)(x2 * rn);
  if (t == 0) m2[row] = tot * rn * rn;
}

// ---------------------------------------------------------------------------
// Stage 3: fused WMMA GEMM + min-over-bank reduction.
// Block: 256 threads = 8 waves; 32 patches (2 A tiles, staged in LDS once and
// re-read per k-step) x 2048 bank columns. Each wave owns two 16-col N-tiles/pass.
// Steady state: 4x v_wmma_f32_16x16x32_f16 per 4x clause-batched
// global_load_b128 from the L2-resident f16 bank.
// ---------------------------------------------------------------------------
__global__ __launch_bounds__(256) void patchcore_nn_kernel(
    const _Float16* __restrict__ A, const _Float16* __restrict__ B,
    const float* __restrict__ p2, const float* __restrict__ m2,
    int* __restrict__ mind2)
{
  __shared__ __align__(16) _Float16 As[PATCHES_PER_BLK * APITCH];
  __shared__ int smin[PATCHES_PER_BLK];

  const int tid  = threadIdx.x;
  const int lane = tid & 31;
  const int wave = tid >> 5;
  const int half = lane >> 4;     // lane half selects K sub-ranges (ISA 7.12.2)
  const int ln   = lane & 15;     // row (A) / column (B,C) within tile

  const int       pbase = blockIdx.x * PATCHES_PER_BLK;
  const long long mbase = (long long)blockIdx.y * CHUNK;

  // Cooperative A-tile load into LDS (32 rows x 768 f16, 16B vectors)
  for (int idx = tid; idx < PATCHES_PER_BLK * (D_DIM / 8); idx += 256) {
    const int r = idx / (D_DIM / 8), c = idx % (D_DIM / 8);
    *(uint4*)&As[r * APITCH + c * 8] =
        *(const uint4*)&A[(long long)(pbase + r) * D_DIM + c * 8];
  }
  if (tid < PATCHES_PER_BLK) smin[tid] = 0x7F800000;
  __syncthreads();

  const float INF = __int_as_float(0x7F800000);
  v8f rq00, rq10, rq01, rq11;      // running min of (m2 - 2*cross)
#pragma unroll
  for (int i = 0; i < 8; ++i) { rq00[i] = INF; rq10[i] = INF; rq01[i] = INF; rq11[i] = INF; }

  const _Float16* a0row = &As[(0 * 16 + ln) * APITCH];   // A tile 0, row M=ln
  const _Float16* a1row = &As[(1 * 16 + ln) * APITCH];   // A tile 1

  for (int pass = 0; pass < CHUNK / 256; ++pass) {       // 8 passes
    const long long n0 = mbase + pass * 256 + wave * 32; // 32 columns / wave
    const long long ca = n0 + ln;                        // N-tile a column
    const long long cb = n0 + 16 + ln;                   // N-tile b column
    const long long ra = (ca < MBANK) ? ca : (MBANK - 1);
    const long long rb = (cb < MBANK) ? cb : (MBANK - 1);
    const _Float16* browa = B + ra * D_DIM + half * 8;   // B frag: N=ln, K chunks at half*8 and half*8+16
    const _Float16* browb = B + rb * D_DIM + half * 8;
    const float m2a = (ca < MBANK) ? m2[ca] : INF;       // OOB columns masked to +INF
    const float m2b = (cb < MBANK) ? m2[cb] : INF;

    v8f c00 = {}, c10 = {}, c01 = {}, c11 = {};
    for (int k = 0; k < KSTEPS; ++k) {
      union { v16h v; v8h h[2]; } af0, af1, bfa, bfb;
      // B fragment: lane = col N, K chunks at half*8 and half*8+16
      bfa.h[0] = *(const v8h*)(browa + k * 32);
      bfa.h[1] = *(const v8h*)(browa + k * 32 + 16);
      bfb.h[0] = *(const v8h*)(browb + k * 32);
      bfb.h[1] = *(const v8h*)(browb + k * 32 + 16);
      // A fragment: lane = row M, K chunks at half*8 and half*8+16
      const int ka0 = k * 32 + half * 8;
      af0.h[0] = *(const v8h*)(a0row + ka0);
      af0.h[1] = *(const v8h*)(a0row + ka0 + 16);
      af1.h[0] = *(const v8h*)(a1row + ka0);
      af1.h[1] = *(const v8h*)(a1row + ka0 + 16);

      c00 = __builtin_amdgcn_wmma_f32_16x16x32_f16(false, af0.v, false, bfa.v, (short)0, c00, false, false);
      c10 = __builtin_amdgcn_wmma_f32_16x16x32_f16(false, af1.v, false, bfa.v, (short)0, c10, false, false);
      c01 = __builtin_amdgcn_wmma_f32_16x16x32_f16(false, af0.v, false, bfb.v, (short)0, c01, false, false);
      c11 = __builtin_amdgcn_wmma_f32_16x16x32_f16(false, af1.v, false, bfb.v, (short)0, c11, false, false);
    }
#pragma unroll
    for (int r = 0; r < 8; ++r) {
      rq00[r] = fminf(rq00[r], fmaf(-2.f, c00[r], m2a));
      rq10[r] = fminf(rq10[r], fmaf(-2.f, c10[r], m2a));
      rq01[r] = fminf(rq01[r], fmaf(-2.f, c01[r], m2b));
      rq11[r] = fminf(rq11[r], fmaf(-2.f, c11[r], m2b));
    }
  }

  // C layout: VGPR r, lane L -> M = r + 8*(L>>4), N = L&15.
  // Min over N = xor-shuffle over lane bits 0..3 (halves stay independent).
#pragma unroll
  for (int t = 0; t < 2; ++t) {
#pragma unroll
    for (int r = 0; r < 8; ++r) {
      float v = (t == 0) ? fminf(rq00[r], rq01[r]) : fminf(rq10[r], rq11[r]);
      v = fminf(v, __shfl_xor(v, 1));
      v = fminf(v, __shfl_xor(v, 2));
      v = fminf(v, __shfl_xor(v, 4));
      v = fminf(v, __shfl_xor(v, 8));
      if (ln == 0) {
        const int pl = t * 16 + half * 8 + r;
        const float d2 = fmaxf(v + p2[pbase + pl], 1e-12f);
        atomicMin(&smin[pl], __float_as_int(d2));   // positive-float bit trick
      }
    }
  }
  __syncthreads();
  if (tid < PATCHES_PER_BLK) atomicMin(&mind2[pbase + tid], smin[tid]);
}

// ---------------------------------------------------------------------------
// Stage 4: per-image score = max over 49 patches of sqrt(min d2)
// ---------------------------------------------------------------------------
__global__ void finalize_kernel(const int* __restrict__ mind2, float* __restrict__ out)
{
  const int b = threadIdx.x;
  if (b >= 64) return;
  float best = 0.0f;
  for (int p = 0; p < 49; ++p) {
    const float d2 = __int_as_float(mind2[b * 49 + p]);
    best = fmaxf(best, sqrtf(d2));
  }
  out[b] = best;
}

// ---------------------------------------------------------------------------
extern "C" void kernel_launch(void* const* d_in, const int* in_sizes, int n_in,
                              void* d_out, int out_size, void* d_ws, size_t ws_size,
                              hipStream_t stream)
{
  (void)in_sizes; (void)n_in; (void)out_size; (void)ws_size;
  const float* tokens = (const float*)d_in[0];   // [64,50,768] f32
  const float* bank   = (const float*)d_in[1];   // [100000,768] f32
  float* out = (float*)d_out;                    // [64] f32

  char* w = (char*)d_ws;
  auto take = [&](size_t bytes) -> char* {
    char* p = w; w += (bytes + 255) & ~(size_t)255; return p;
  };
  _Float16* Ah   = (_Float16*)take((size_t)NPATCH * D_DIM * 2);  // 4.8 MB
  _Float16* Bh   = (_Float16*)take((size_t)MBANK  * D_DIM * 2);  // 153.6 MB (L2-resident)
  float*    p2   = (float*)take((size_t)NPATCH * 4);
  float*    m2   = (float*)take((size_t)MBANK  * 4);
  int*      mind2= (int*)take((size_t)NPATCH * 4);

  prep_patches_kernel<<<NPATCH, 256, 0, stream>>>(tokens, Ah, p2, mind2);
  prep_bank_kernel<<<MBANK, 256, 0, stream>>>(bank, Bh, m2);

  dim3 grid(NPATCH / PATCHES_PER_BLK, (MBANK + CHUNK - 1) / CHUNK);  // 98 x 49
  patchcore_nn_kernel<<<grid, 256, 0, stream>>>(Ah, Bh, p2, m2, mind2);

  finalize_kernel<<<1, 64, 0, stream>>>(mind2, out);
}
